// AttentionGuidedPooling_75230647157184
// MI455X (gfx1250) — matrix operand order repaired and checked by reading, not compiled, not run on hardware
//
#include <hip/hip_runtime.h>
#include <stdint.h>

// ---------------------------------------------------------------------------
// AttentionGuidedPooling, MI455X (gfx1250).
// B=4, C=256, H=W=64 -> N=4096, D=256.
// Stage 1: Q/K/V 1x1 convs as bf16 WMMA GEMMs (V stored transposed).
// Stage 2: fused flash attention, bf16 WMMA, double-buffered TDM tile loads.
// ---------------------------------------------------------------------------

#define BB 4
#define CC 256
#define NN 4096
#define DD 256

typedef __attribute__((ext_vector_type(16))) __bf16    v16bf;
typedef __attribute__((ext_vector_type(8)))  float     v8f;
typedef __attribute__((ext_vector_type(4)))  uint32_t  v4u32;
typedef __attribute__((ext_vector_type(4)))  int       v4i32;
typedef __attribute__((ext_vector_type(8)))  int       v8i32;

union Frag {            // one 16x32 bf16 WMMA operand (8 VGPRs)
  v16bf    bf;
  v4u32    q[2];
  uint32_t u[8];
};

// Native bf16 converts (v_cvt_bf16_f32 / v_cvt_pk_bf16_f32), RNE.
static __device__ __forceinline__ uint16_t f2bf(float f) {
  union { __bf16 h; uint16_t u; } p;
  p.h = (__bf16)f;
  return p.u;
}
static __device__ __forceinline__ uint32_t pk2bf(float lo, float hi) {
  union { __bf16 h[2]; uint32_t u; } p;
  p.h[0] = (__bf16)lo;
  p.h[1] = (__bf16)hi;
  return p.u;
}

static __device__ __forceinline__ v8f vzero8() {
  v8f z = {0.f, 0.f, 0.f, 0.f, 0.f, 0.f, 0.f, 0.f};
  return z;
}

#define WMMA_BF16(A, B, C) \
  __builtin_amdgcn_wmma_f32_16x16x32_bf16(false, (A), false, (B), (short)0, (C), false, false)

// ---------------------------------------------------------------------------
// Tensor Data Mover: 2D tile load global->LDS with LDS row padding.
// ---------------------------------------------------------------------------
#if __has_include(<hip/amd_detail/amd_gfx1250_TDM.h>)
#define TDM_6ARG 1
#endif

static __device__ __forceinline__ void tdm_load_2d(
    uint32_t lds_off, const void* gptr,
    uint32_t tensor_d0, uint32_t tensor_d1,
    uint32_t tile_d0, uint32_t tile_d1,
    uint64_t stride0_elems,
    uint32_t pad_en, uint32_t pad_interval_code, uint32_t pad_amount_code)
{
  uint64_t ga = (uint64_t)(uintptr_t)gptr;
  v4u32 g0;
  g0[0] = 1u;                                     // count=1, user mode, no gather
  g0[1] = lds_off;                                // lds_addr
  g0[2] = (uint32_t)ga;                           // global_addr[31:0]
  g0[3] = (uint32_t)((ga >> 32) & 0x01FFFFFFu) | (2u << 30);  // addr[56:32], type=2

  uint32_t w0 = (1u << 16)                        // data_size = 2 bytes (bf16)
              | (pad_en << 20)
              | (pad_interval_code << 22)
              | (pad_amount_code << 25);
  v8i32 g1;
  g1[0] = (int)w0;
  g1[1] = (int)((tensor_d0 & 0xFFFFu) << 16);               // barrier_addr=0 | dim0.lo
  g1[2] = (int)((tensor_d0 >> 16) | ((tensor_d1 & 0xFFFFu) << 16));
  g1[3] = (int)((tensor_d1 >> 16) | ((tile_d0 & 0xFFFFu) << 16));
  g1[4] = (int)(tile_d1 & 0xFFFFu);                         // tile_dim2 = 0
  g1[5] = (int)(uint32_t)(stride0_elems & 0xFFFFFFFFu);
  g1[6] = (int)(uint32_t)((stride0_elems >> 32) & 0xFFFFu); // dim1_stride = 0 (2D)
  g1[7] = 0;

  v4i32 z4 = {0, 0, 0, 0};
#ifdef TDM_6ARG
  v8i32 z8 = {0, 0, 0, 0, 0, 0, 0, 0};
  __builtin_amdgcn_tensor_load_to_lds(g0, g1, z4, z4, z8, 0);
#else
  __builtin_amdgcn_tensor_load_to_lds(g0, g1, z4, z4, 0);
#endif
}

// ---------------------------------------------------------------------------
// Stage 1: Y = W(256x256) @ X(256x4096) + bias, stored bf16.
//   which 0: Q <- tgt (row-major o,p)    [flat buffer == Qm(4096x256)]
//   which 1: K <- src (row-major o,p)
//   which 2: V <- src, stored TRANSPOSED: Vt[c*4096 + m], m = 16*o + (p>>8), c = p&255
// blockIdx: x = p-tile (64 wide), y = o-tile (64 tall), z = which*4 + b
// ---------------------------------------------------------------------------
#define S1_ROW 40   // LDS Xt row stride in bf16 elems (32 + pad), 16B-aligned chunks

__global__ __launch_bounds__(128)
void qkv_gemm(const float* __restrict__ tgt, const float* __restrict__ src,
              const float* __restrict__ qw, const float* __restrict__ qb,
              const float* __restrict__ kw, const float* __restrict__ kb,
              const float* __restrict__ vw, const float* __restrict__ vb,
              uint16_t* __restrict__ qbuf, uint16_t* __restrict__ kbuf,
              uint16_t* __restrict__ vtbuf)
{
  __shared__ __align__(16) uint16_t ldsX[64 * S1_ROW];

  const int z = blockIdx.z;
  const int b = z & 3;
  const int which = z >> 2;

  const float* x    = (which == 0) ? tgt : src;
  const float* w    = (which == 0) ? qw : (which == 1) ? kw : vw;
  const float* bias = (which == 0) ? qb : (which == 1) ? kb : vb;
  uint16_t* dst     = (which == 0) ? qbuf : (which == 1) ? kbuf : vtbuf;

  const int tid  = threadIdx.x;
  const int wave = tid >> 5;
  const int lane = tid & 31;
  const int l16  = lane & 15;
  const int hi   = lane >> 4;            // 0: lanes 0-15, 1: lanes 16-31

  const float* xb = x + (size_t)b * CC * NN;
  const int p0 = blockIdx.x * 64;
  const int o0 = blockIdx.y * 64 + wave * 16;

  v8f acc[4] = {vzero8(), vzero8(), vzero8(), vzero8()};

  for (int k0 = 0; k0 < CC; k0 += 32) {
    // ---- cooperative fp32->bf16 transpose of X[k0..k0+32)[p0..p0+64) into LDS
    // thread: ci row-pair (2ip, 2ip+1) x 8 cols; packed b32 LDS stores
    {
      const int ip = tid >> 3;             // 0..15 -> ci rows 2ip, 2ip+1
      const int cg = (tid & 7) * 8;        // p col group
      const float* r0 = xb + (size_t)(k0 + 2 * ip) * NN + p0 + cg;
      const float* r1 = r0 + NN;
#pragma unroll
      for (int v = 0; v < 2; ++v) {
        float4 f0 = *(const float4*)(r0 + 4 * v);
        float4 f1 = *(const float4*)(r1 + 4 * v);
        *(uint32_t*)&ldsX[(cg + 4 * v + 0) * S1_ROW + 2 * ip] = pk2bf(f0.x, f1.x);
        *(uint32_t*)&ldsX[(cg + 4 * v + 1) * S1_ROW + 2 * ip] = pk2bf(f0.y, f1.y);
        *(uint32_t*)&ldsX[(cg + 4 * v + 2) * S1_ROW + 2 * ip] = pk2bf(f0.z, f1.z);
        *(uint32_t*)&ldsX[(cg + 4 * v + 3) * S1_ROW + 2 * ip] = pk2bf(f0.w, f1.w);
      }
    }
    __syncthreads();

    // ---- A fragment: W rows (fp32 -> bf16), ISA 16-bit A layout:
    // lane<16: K {0..7, 16..23};  lane>=16: K {8..15, 24..31}
    Frag a;
    {
      const float* wr = w + (size_t)(o0 + l16) * CC + k0;
      const int c0 = hi * 8;
#pragma unroll
      for (int e = 0; e < 4; ++e) {
        a.u[e]     = pk2bf(wr[c0 + 2 * e],      wr[c0 + 2 * e + 1]);
        a.u[4 + e] = pk2bf(wr[16 + c0 + 2 * e], wr[16 + c0 + 2 * e + 1]);
      }
    }

    // ---- B fragments from LDS; same K order as A: lane<16 K {0..7, 16..23}, lane>=16 K {8..15, 24..31}
#pragma unroll
    for (int ns = 0; ns < 4; ++ns) {
      Frag bf_;
      const uint16_t* row = &ldsX[(ns * 16 + l16) * S1_ROW + hi * 8];
      bf_.q[0] = *(const v4u32*)(row);
      bf_.q[1] = *(const v4u32*)(row + 16);
      acc[ns] = WMMA_BF16(a.bf, bf_.bf, acc[ns]);
    }
    __syncthreads();
  }

  // ---- epilogue: + bias, cvt bf16, store (C layout: VGPR j = rows j / j+8)
#pragma unroll
  for (int j = 0; j < 8; ++j) {
    const int o = o0 + j + hi * 8;
    const float bv = bias[o];
#pragma unroll
    for (int ns = 0; ns < 4; ++ns) {
      const float v = acc[ns][j] + bv;
      const int p = p0 + ns * 16 + l16;
      size_t idx;
      if (which == 2) {
        const int c = p & 255, r = p >> 8;
        idx = ((size_t)b * DD + c) * NN + 16 * o + r;     // Vt[c][m]
      } else {
        idx = ((size_t)b * CC + o) * NN + p;              // Y row-major == (n,c)
      }
      dst[idx] = f2bf(v);
    }
  }
}

// ---------------------------------------------------------------------------
// Stage 2: flash attention.  Per batch: Qm,Km (4096x256), Vt (256x4096).
// Block = 4 waves, each wave owns 16 query rows; 64-key tiles, double-buffered
// TDM DMA into LDS overlapped with WMMA compute.
// ---------------------------------------------------------------------------
#define KT   64
#define NT   (NN / KT)
#define KROW 264   // 256 + 8 elems (TDM pads 4 dwords every 128 dwords)
#define VROW 72    // 64 + 8 elems  (TDM pads 4 dwords every 32 dwords)
#define PROW 72

__global__ __launch_bounds__(128)
void flash_attn(const uint16_t* __restrict__ qbuf,
                const uint16_t* __restrict__ kbuf,
                const uint16_t* __restrict__ vtbuf,
                float* __restrict__ out)
{
  __shared__ __align__(16) uint16_t ldsK[2][KT * KROW];     // 2 x 33 KB
  __shared__ __align__(16) uint16_t ldsV[2][DD * VROW];     // 2 x 36 KB
  __shared__ __align__(16) uint16_t ldsP[4 * 16 * PROW];    //     9 KB

  const int b    = blockIdx.y;
  const int rb   = blockIdx.x;
  const int tid  = threadIdx.x;
  const int wave = tid >> 5;
  const int lane = tid & 31;
  const int l16  = lane & 15;
  const int hi   = lane >> 4;
  const int n0   = rb * 64 + wave * 16;

  const uint16_t* kbase = kbuf + (size_t)b * NN * DD;
  const uint16_t* vbase = vtbuf + (size_t)b * DD * NN;

  // Q resident: 8 A-fragments covering D=256
  Frag qf[8];
  {
    const uint16_t* qrow = qbuf + ((size_t)b * NN + (n0 + l16)) * DD;
#pragma unroll
    for (int kc = 0; kc < 8; ++kc) {
      const uint16_t* pch = qrow + kc * 32 + hi * 8;
      qf[kc].q[0] = *(const v4u32*)(pch);
      qf[kc].q[1] = *(const v4u32*)(pch + 16);
    }
  }

  v8f accO[16];
#pragma unroll
  for (int i = 0; i < 16; ++i) accO[i] = vzero8();
  float m8[8], l8[8];
#pragma unroll
  for (int j = 0; j < 8; ++j) { m8[j] = -3.0e38f; l8[j] = 0.f; }

  // ---- prologue: DMA tile 0 into buffer 0
  if (wave == 0) {
    tdm_load_2d((uint32_t)(uintptr_t)&ldsK[0][0], kbase,
                DD, KT, DD, KT, DD, 1, /*128dw*/6, /*4dw*/3);
    tdm_load_2d((uint32_t)(uintptr_t)&ldsV[0][0], vbase,
                KT, DD, KT, DD, NN, 1, /*32dw*/4, /*4dw*/3);
  }

  for (int t = 0; t < NT; ++t) {
    if (wave == 0) __builtin_amdgcn_s_wait_tensorcnt(0);   // tile t landed
    __syncthreads();   // tile t visible; all waves done reading tile t-1

    // ---- prefetch tile t+1 into the other buffer (overlaps compute below)
    if (wave == 0 && t + 1 < NT) {
      const int m1 = (t + 1) * KT;
      const int bi = (t + 1) & 1;
      tdm_load_2d((uint32_t)(uintptr_t)&ldsK[bi][0], kbase + (size_t)m1 * DD,
                  DD, KT, DD, KT, DD, 1, 6, 3);
      tdm_load_2d((uint32_t)(uintptr_t)&ldsV[bi][0], vbase + m1,
                  KT, DD, KT, DD, NN, 1, 4, 3);
    }

    const uint16_t* Kb = ldsK[t & 1];
    const uint16_t* Vb = ldsV[t & 1];

    // ---- S = Q K^T : four 16x16 subtiles over 64 keys
    v8f s[4];
#pragma unroll
    for (int ns = 0; ns < 4; ++ns) {
      v8f a = vzero8();
#pragma unroll
      for (int kc = 0; kc < 8; ++kc) {
        Frag bf_;
        const uint16_t* kr = &Kb[(ns * 16 + l16) * KROW + kc * 32 + hi * 8];
        bf_.q[0] = *(const v4u32*)(kr);
        bf_.q[1] = *(const v4u32*)(kr + 16);
        a = WMMA_BF16(qf[kc].bf, bf_.bf, a);
      }
      s[ns] = a;
    }

    // ---- online softmax (row r = j + hi*8; reduce across 16-lane half)
    float alpha[8];
#pragma unroll
    for (int j = 0; j < 8; ++j) {
      float tm = fmaxf(fmaxf(s[0][j], s[1][j]), fmaxf(s[2][j], s[3][j]));
#pragma unroll
      for (int d = 1; d < 16; d <<= 1) tm = fmaxf(tm, __shfl_xor(tm, d, 32));
      const float mnew = fmaxf(m8[j], tm);
      alpha[j] = __expf(m8[j] - mnew);
      m8[j] = mnew;
      float rs = 0.f;
#pragma unroll
      for (int ns = 0; ns < 4; ++ns) {
        const float p = __expf(s[ns][j] - mnew);
        s[ns][j] = p;
        rs += p;
      }
#pragma unroll
      for (int d = 1; d < 16; d <<= 1) rs += __shfl_xor(rs, d, 32);
      l8[j] = l8[j] * alpha[j] + rs;
    }

    // ---- P -> bf16 -> per-wave LDS scratch; rescale O
    uint16_t* pbase = &ldsP[wave * 16 * PROW];
#pragma unroll
    for (int j = 0; j < 8; ++j) {
      const int r = j + hi * 8;
#pragma unroll
      for (int ns = 0; ns < 4; ++ns)
        pbase[r * PROW + ns * 16 + l16] = f2bf(s[ns][j]);
#pragma unroll
      for (int df = 0; df < 16; ++df) accO[df][j] *= alpha[j];
    }

    // ---- reload P as A fragments (2 chunks of K=32 over the 64 keys)
    Frag pf[2];
#pragma unroll
    for (int kf = 0; kf < 2; ++kf) {
      const uint16_t* pr = pbase + l16 * PROW + kf * 32 + hi * 8;
      pf[kf].q[0] = *(const v4u32*)(pr);
      pf[kf].q[1] = *(const v4u32*)(pr + 16);
    }

    // ---- O += P * V  (B operand = Vt rows, contiguous in m)
#pragma unroll
    for (int df = 0; df < 16; ++df) {
#pragma unroll
      for (int kf = 0; kf < 2; ++kf) {
        Frag bf_;
        const uint16_t* vr = &Vb[(df * 16 + l16) * VROW + kf * 32 + hi * 8];
        bf_.q[0] = *(const v4u32*)(vr);
        bf_.q[1] = *(const v4u32*)(vr + 16);
        accO[df] = WMMA_BF16(pf[kf].bf, bf_.bf, accO[df]);
      }
    }
  }

  // ---- normalize and store fp32 (out flat == (B,N,C) row-major)
  float* ob = out + (size_t)b * NN * DD;
#pragma unroll
  for (int j = 0; j < 8; ++j) {
    const float inv = 1.f / l8[j];
    const int n = n0 + j + hi * 8;
#pragma unroll
    for (int df = 0; df < 16; ++df)
      ob[(size_t)n * DD + df * 16 + l16] = accO[df][j] * inv;
  }
}

// ---------------------------------------------------------------------------
extern "C" void kernel_launch(void* const* d_in, const int* in_sizes, int n_in,
                              void* d_out, int out_size, void* d_ws, size_t ws_size,
                              hipStream_t stream) {
  const float* tgt = (const float*)d_in[0];
  const float* src = (const float*)d_in[1];
  const float* qw  = (const float*)d_in[2];
  const float* qb  = (const float*)d_in[3];
  const float* kw  = (const float*)d_in[4];
  const float* kb  = (const float*)d_in[5];
  const float* vw  = (const float*)d_in[6];
  const float* vb  = (const float*)d_in[7];

  uint16_t* qbuf  = (uint16_t*)d_ws;                       // 8 MB
  uint16_t* kbuf  = qbuf + (size_t)BB * NN * DD;           // 8 MB
  uint16_t* vtbuf = kbuf + (size_t)BB * NN * DD;           // 8 MB

  dim3 g1(NN / 64, CC / 64, 3 * BB);
  qkv_gemm<<<g1, 128, 0, stream>>>(tgt, src, qw, qb, kw, kb, vw, vb,
                                   qbuf, kbuf, vtbuf);

  dim3 g2(NN / 64, BB);
  flash_attn<<<g2, 128, 0, stream>>>(qbuf, kbuf, vtbuf, (float*)d_out);
}
